// TestModule_22874995818886
// MI455X (gfx1250) — matrix operand order repaired and checked by reading, not compiled, not run on hardware
//
#include <hip/hip_runtime.h>
#include <hip/hip_bf16.h>

// Problem constants (match reference)
#define BATCH   4
#define SEQ     1024
#define NTOK    (BATCH * SEQ)      // 4096 rows (M)
#define D_EMB   1024
#define A_PAD   64
#define KDIM    (D_EMB + A_PAD)    // 1088
#define NDIM    32000              // VOCAB_OUT
#define MDIM    NTOK
#define KSYN    8

typedef _Float16 v16h __attribute__((ext_vector_type(16)));
typedef _Float16 v8h  __attribute__((ext_vector_type(8)));
typedef float    v8f  __attribute__((ext_vector_type(8)));

union HVec { v16h v; v8h h[2]; };

// ---------------------------------------------------------------------------
// CDNA5 Tensor Data Mover support (6-arg builtin: clang-23 / therock form)
// ---------------------------------------------------------------------------
#if defined(__has_builtin)
# if __has_builtin(__builtin_amdgcn_tensor_load_to_lds) && \
     __has_builtin(__builtin_amdgcn_s_wait_tensorcnt)
#  define USE_TDM 1
# endif
#endif
#ifndef USE_TDM
# define USE_TDM 0
#endif

#if USE_TDM
typedef unsigned int u32x4 __attribute__((ext_vector_type(4)));
typedef int          i32x8 __attribute__((ext_vector_type(8)));
typedef int          i32x4 __attribute__((ext_vector_type(4)));

// Issue one TDM 2-D tile load: tile_dim0=32 halves, tile_dim1=128 rows,
// row stride KDIM halves in memory; LDS rows padded 64B+32B -> 96B stride
// (pad_interval=3 -> every 16 DWORDs, pad_amount=7 -> 8 DWORDs).
__device__ inline void tdm_load_tile(unsigned long long gaddr,
                                     unsigned lds_addr, unsigned rows) {
    u32x4 g0;
    g0.x = 1u;                                   // count=1, user mode, no gather
    g0.y = lds_addr;                             // D# bits 63:32  lds_addr
    g0.z = (unsigned)gaddr;                      // D# bits 95:64  global_addr lo
    g0.w = (unsigned)(gaddr >> 32) | (2u << 30); // addr[56:32] | type=2 (image)
    i32x8 g1;
    g1[0] = (1 << 16)                            // data_size = 1 -> 2 bytes
          | (1 << 20)                            // pad_enable
          | (3 << 22)                            // pad_interval: 16 DWORDs (64 B)
          | (7 << 25);                           // pad_amount:   8 DWORDs (32 B)
    g1[1] = (int)((unsigned)KDIM << 16);         // tensor_dim0[15:0] @ bits 63:48
    g1[2] = (int)(rows << 16);                   // dim0 hi=0 | tensor_dim1[15:0]
    g1[3] = (int)((rows >> 16) | (32u << 16));   // dim1 hi | tile_dim0 = 32
    g1[4] = 128;                                 // tile_dim1 = 128, tile_dim2 = 0
    g1[5] = KDIM;                                // tensor_dim0_stride (lo 32)
    g1[6] = 0;                                   // stride hi | dim1_stride lo
    g1[7] = 0;
    i32x4 z4 = {0, 0, 0, 0};                     // groups 2/3: unused (2-D tensor)
    i32x8 z8 = {0, 0, 0, 0, 0, 0, 0, 0};
    __builtin_amdgcn_tensor_load_to_lds(g0, g1, z4, z4, z8, 0);
}
#endif

// ---------------------------------------------------------------------------
// Block reduction (LDS tree, wave32-safe: no hardcoded wave width)
// ---------------------------------------------------------------------------
__device__ inline float blockReduceSum(float v, float* red) {
    const int tid = threadIdx.x;
    red[tid] = v;
    __syncthreads();
    for (int off = 128; off > 0; off >>= 1) {
        if (tid < off) red[tid] += red[tid + off];
        __syncthreads();
    }
    float r = red[0];
    __syncthreads();   // protect red[] for the next reduction
    return r;
}

// ---------------------------------------------------------------------------
// Kernel 1: convert W_rev [K,N] fp32 -> WT [N,K] f16 (LDS-tiled transpose).
// Grid: (NDIM/64, KDIM/64) = (500, 17), block 256.
// ---------------------------------------------------------------------------
__global__ void __launch_bounds__(256)
transpose_wrev(const float* __restrict__ Wrev, _Float16* __restrict__ WT) {
    __shared__ _Float16 tile[64][65];
    const int kBase = blockIdx.y * 64;
    const int nBase = blockIdx.x * 64;
    const int tn = threadIdx.x & 63;   // fast dim for coalesced global read
    const int tk = threadIdx.x >> 6;   // 0..3

    #pragma unroll
    for (int r = 0; r < 16; ++r) {
        const int kk = tk + r * 4;
        tile[kk][tn] = (_Float16)Wrev[(size_t)(kBase + kk) * NDIM + nBase + tn];
    }
    __syncthreads();
    #pragma unroll
    for (int r = 0; r < 16; ++r) {
        const int nn = tk + r * 4;
        WT[(size_t)(nBase + nn) * KDIM + kBase + tn] = tile[tn][nn];
    }
}

// ---------------------------------------------------------------------------
// Kernel 2: VirtualEmbedding -> Mid [NTOK, KDIM] f16 (incl. padding cols).
// One block per token; 8 synonym rows cached in LDS (32 KB of 320 KB WGP LDS).
// Grid: NTOK blocks, 256 threads.
// ---------------------------------------------------------------------------
__global__ void __launch_bounds__(256)
build_mid(const int* __restrict__ ids,
          const int* __restrict__ syn_table,
          const unsigned char* __restrict__ syn_mask,
          const float* __restrict__ W_emb,
          const float* __restrict__ padding,
          _Float16* __restrict__ Mid) {
    __shared__ float synv[KSYN][D_EMB];   // 32 KB
    __shared__ float red[256];

    const int t   = blockIdx.x;
    const int tid = threadIdx.x;
    const int id  = ids[t];

    // base embedding row sum
    float s = 0.f;
    for (int d = tid; d < D_EMB; d += 256) s += W_emb[(size_t)id * D_EMB + d];
    const float i_sum = blockReduceSum(s, red);

    // synonym rows: stage in LDS, reduce per-row sums, compute scales
    float scale[KSYN];
    #pragma unroll
    for (int k = 0; k < KSYN; ++k) {
        const int sid = syn_table[id * KSYN + k];
        float ps = 0.f;
        for (int d = tid; d < D_EMB; d += 256) {
            const float v = W_emb[(size_t)sid * D_EMB + d];
            synv[k][d] = v;
            ps += v;
        }
        const float denom = blockReduceSum(ps, red);
        scale[k] = syn_mask[id * KSYN + k] ? (i_sum / denom) : 0.f;
    }

    // weighted combine (each thread reads back only the LDS slots it wrote)
    for (int d = tid; d < D_EMB; d += 256) {
        float acc = 0.f;
        #pragma unroll
        for (int k = 0; k < KSYN; ++k) acc += scale[k] * synv[k][d];
        Mid[(size_t)t * KDIM + d] = (_Float16)acc;
    }
    if (tid < A_PAD)
        Mid[(size_t)t * KDIM + D_EMB + tid] = (_Float16)padding[t * A_PAD + tid];
}

// ---------------------------------------------------------------------------
// Kernel 3: GEMM  out[M,N] = Mid[M,K](f16) * WT[N,K](f16, B pre-transposed).
// Block tile 128x128, BK=32, 256 threads = 8 waves, each wave 2x4 WMMA tiles.
// Tile staging: double-buffered TENSOR_LOAD_TO_LDS (TDM) when available,
// manual global->LDS copies otherwise. Grid: (NDIM/128, MDIM/128).
// ---------------------------------------------------------------------------
__global__ void __launch_bounds__(256)
gemm_f16_wmma(const _Float16* __restrict__ Mid,
              const _Float16* __restrict__ WT,
              float* __restrict__ out) {
    constexpr int BK  = 32;
    constexpr int LDA = 48;   // halves/row in LDS: 96 B rows, 16-byte aligned
    // [buffer][A|B][128 rows * LDA halves] : 48 KB total
    __shared__ __align__(16) _Float16 smem[2][2][128 * LDA];

    const int tid  = threadIdx.x;
    const int lane = tid & 31;
    const int wave = tid >> 5;
    const int wm   = wave & 3;     // M quadrant: rows 32*wm .. +31
    const int wn   = wave >> 2;    // N half:    cols 64*wn .. +63
    const int l16  = lane & 15;
    const int hi   = lane >> 4;

    const int mBase = blockIdx.y * 128;
    const int nBase = blockIdx.x * 128;

    const v8f zero = {};
    v8f acc[2][4];
    #pragma unroll
    for (int i = 0; i < 2; ++i)
        #pragma unroll
        for (int j = 0; j < 4; ++j) acc[i][j] = zero;

#if USE_TDM
    const unsigned long long gA =
        (unsigned long long)(size_t)(Mid + (size_t)mBase * KDIM);
    const unsigned long long gB =
        (unsigned long long)(size_t)(WT + (size_t)nBase * KDIM);
    // prologue: stage K-step 0 into buffer 0 (wave 0 drives the TDM)
    if (wave == 0) {
        tdm_load_tile(gA, (unsigned)(size_t)&smem[0][0][0], MDIM);
        tdm_load_tile(gB, (unsigned)(size_t)&smem[0][1][0], NDIM);
    }
    int cur = 0;
#endif

    for (int k0 = 0; k0 < KDIM; k0 += BK) {
#if USE_TDM
        if (wave == 0) {
            if (k0 + BK < KDIM) {
                // issue next tile into the alternate buffer, then wait for the
                // PREVIOUS pair only (TDM ops retire in order per wave)
                const unsigned long long koff =
                    (unsigned long long)(k0 + BK) * sizeof(_Float16);
                tdm_load_tile(gA + koff,
                              (unsigned)(size_t)&smem[cur ^ 1][0][0], MDIM);
                tdm_load_tile(gB + koff,
                              (unsigned)(size_t)&smem[cur ^ 1][1][0], NDIM);
                __builtin_amdgcn_s_wait_tensorcnt(2);
            } else {
                __builtin_amdgcn_s_wait_tensorcnt(0);
            }
        }
        __syncthreads();
        const _Float16* As = smem[cur][0];
        const _Float16* Bs = smem[cur][1];
#else
        // manual staging: 128 rows x 32 halves each side, v8h (16 B) moves
        #pragma unroll
        for (int r = 0; r < 2; ++r) {
            const int li  = tid + r * 256;
            const int row = li >> 2;
            const int c8  = (li & 3) * 8;
            *(v8h*)(&smem[0][0][0] + row * LDA + c8) =
                *(const v8h*)(Mid + (size_t)(mBase + row) * KDIM + k0 + c8);
            *(v8h*)(&smem[0][1][0] + row * LDA + c8) =
                *(const v8h*)(WT + (size_t)(nBase + row) * KDIM + k0 + c8);
        }
        __syncthreads();
        const _Float16* As = smem[0][0];
        const _Float16* Bs = smem[0][1];
#endif

        // A fragment: lane m = l16; halves e<8 -> K=hi*8+e, e>=8 -> K=16+hi*8+(e-8)
        HVec a[2];
        #pragma unroll
        for (int i = 0; i < 2; ++i) {
            const int m = 32 * wm + 16 * i + l16;
            a[i].h[0] = *(const v8h*)(As + m * LDA + hi * 8);
            a[i].h[1] = *(const v8h*)(As + m * LDA + 16 + hi * 8);
        }
        // B fragment: lane n = l16; halves e<8 -> K=hi*8+e, e>=8 -> K=16+hi*8+(e-8)
        HVec b[4];
        #pragma unroll
        for (int j = 0; j < 4; ++j) {
            const int n = 64 * wn + 16 * j + l16;
            b[j].h[0] = *(const v8h*)(Bs + n * LDA + hi * 8);
            b[j].h[1] = *(const v8h*)(Bs + n * LDA + 16 + hi * 8);
        }

        #pragma unroll
        for (int i = 0; i < 2; ++i)
            #pragma unroll
            for (int j = 0; j < 4; ++j)
                acc[i][j] = __builtin_amdgcn_wmma_f32_16x16x32_f16(
                    false, a[i].v, false, b[j].v,
                    (short)0, acc[i][j], false, false);

        __syncthreads();   // buffer fully consumed before it is re-targeted
#if USE_TDM
        cur ^= 1;
#endif
    }

    // D layout: VGPR v holds row m = v + 8*hi, col n = l16
    #pragma unroll
    for (int i = 0; i < 2; ++i) {
        #pragma unroll
        for (int j = 0; j < 4; ++j) {
            const int col = nBase + 64 * wn + 16 * j + l16;
            #pragma unroll
            for (int v = 0; v < 8; ++v) {
                const int row = mBase + 32 * wm + 16 * i + v + 8 * hi;
                out[(size_t)row * NDIM + col] = acc[i][j][v];
            }
        }
    }
}

// ---------------------------------------------------------------------------
// Launch
// ---------------------------------------------------------------------------
extern "C" void kernel_launch(void* const* d_in, const int* in_sizes, int n_in,
                              void* d_out, int out_size, void* d_ws, size_t ws_size,
                              hipStream_t stream) {
    const int*           ids       = (const int*)d_in[0];
    const int*           syn_table = (const int*)d_in[1];
    const unsigned char* syn_mask  = (const unsigned char*)d_in[2];
    const float*         W_emb     = (const float*)d_in[3];
    const float*         W_rev     = (const float*)d_in[4];
    const float*         padding   = (const float*)d_in[5];
    float*               out       = (float*)d_out;

    // workspace: Mid f16 [M,K] (8.9 MB) then WT f16 [N,K] (69.6 MB)
    _Float16* Mid = (_Float16*)d_ws;
    _Float16* WT  = (_Float16*)((char*)d_ws + (size_t)MDIM * KDIM * sizeof(_Float16));

    transpose_wrev<<<dim3(NDIM / 64, KDIM / 64), 256, 0, stream>>>(W_rev, WT);
    build_mid<<<dim3(NTOK), 256, 0, stream>>>(ids, syn_table, syn_mask, W_emb,
                                              padding, Mid);
    gemm_f16_wmma<<<dim3(NDIM / 128, MDIM / 128), 256, 0, stream>>>(Mid, WT, out);
}
